// UNQ_5162550690347
// MI455X (gfx1250) — matrix operand order repaired and checked by reading, not compiled, not run on hardware
//
#include <hip/hip_runtime.h>
#include <hip/hip_bf16.h>

// ---------------------------------------------------------------------------
// Problem constants
// ---------------------------------------------------------------------------
#define NROWS 16384   // B*T
#define DIMX  512
#define HIDX  1024
#define QN    8
#define CN    256
#define DN    256
#define EPSB  1e-5f

typedef __bf16 bf16;
typedef __attribute__((ext_vector_type(16))) __bf16 v16bf;
typedef __attribute__((ext_vector_type(8)))  __bf16 v8bf;
typedef __attribute__((ext_vector_type(4)))  __bf16 v4bf;
typedef __attribute__((ext_vector_type(8)))  float  v8f;

union Frag16 { v16bf v; v8bf h[2]; };

// ---------------------------------------------------------------------------
// Transpose x[B, DIM, T] -> xt[(b*T+t), DIM]
// ---------------------------------------------------------------------------
__global__ __launch_bounds__(256) void k_transpose(const float* __restrict__ x,
                                                   float* __restrict__ xt) {
  __shared__ float tile[32][33];
  const int b  = blockIdx.z;
  const int d0 = blockIdx.y * 32;
  const int t0 = blockIdx.x * 32;
  const int tx = threadIdx.x & 31;
  const int ty = threadIdx.x >> 5;  // 0..7
#pragma unroll
  for (int i = 0; i < 4; ++i) {
    int d = ty + i * 8;
    tile[d][tx] = x[((long)b * DIMX + (d0 + d)) * 512 + (t0 + tx)];
  }
  __syncthreads();
#pragma unroll
  for (int i = 0; i < 4; ++i) {
    int t = ty + i * 8;
    xt[((long)(b * 512 + t0 + t)) * DIMX + (d0 + tx)] = tile[tx][t];
  }
}

// ---------------------------------------------------------------------------
// Generic WMMA bf16 GEMM:   Y[N x Ncols] = pre(A)[N x K] @ W[Ncols x K]^T + b
//   PRE = 0: A as-is
//   PRE = 1: A' = A*scale[k] + shift[k]            (fused BatchNorm)
//   PRE = 2: A' = relu(A*scale[k] + shift[k])      (fused BN + ReLU)
//   LOSS   : also accumulate sum((Y - Xref)^2) into per-block partials
// Block tile 128x128x32, 256 threads = 8 waves, wave tile 32x64 (2x4 WMMA).
// Double-buffered LDS, one barrier per k-step, batched global loads that
// overlap with WMMA of the previous step.
// ---------------------------------------------------------------------------
#define BM 128
#define BN_TILE 128
#define BK 32
#define BKP 48  // padded LDS row (halves): 96B stride, keeps b128 alignment

template <int PRE, bool LOSS>
__global__ __launch_bounds__(256) void k_gemm_wmma(
    const float* __restrict__ A, long lda,
    const float* __restrict__ scale, const float* __restrict__ shift,
    const float* __restrict__ W,       // [Ncols x K], row stride K
    const float* __restrict__ bias,    // may be nullptr
    float* __restrict__ Y, long ldy,
    int K, int Ncols,
    const float* __restrict__ Xref,    // row stride == Ncols (LOSS only)
    float* __restrict__ lossPartial) {
  __shared__ bf16 lds_a[2][BM * BKP];
  __shared__ bf16 lds_b[2][BN_TILE * BKP];
  __shared__ float red[256];

  const int tid     = threadIdx.x;
  const int lane    = tid & 31;
  const int wave    = tid >> 5;
  const int wm      = wave & 3;   // 4 waves along M (32 rows each)
  const int wn      = wave >> 2;  // 2 waves along N (64 cols each)
  const int lane_lo = lane & 15;
  const int lane_hi = lane >> 4;  // 0|1
  const int row_blk = blockIdx.y * BM;
  const int col_blk = blockIdx.x * BN_TILE;

  // staging addressing: f = tid + i*256 -> row = (tid>>3)+i*32, kq const
  const int r0 = tid >> 3;
  const int kq = (tid & 7) << 2;
  const float* Abase = A + (long)(row_blk + r0) * lda + kq;
  const float* Wbase = W + (long)(col_blk + r0) * K + kq;
  const long  astep = 32l * lda;   // row stride per unrolled i
  const long  wstep = 32l * K;

  v8f acc[2][4];
#pragma unroll
  for (int mi = 0; mi < 2; ++mi)
#pragma unroll
    for (int ni = 0; ni < 4; ++ni)
#pragma unroll
      for (int e = 0; e < 8; ++e) acc[mi][ni][e] = 0.0f;

  float4 ra[4], rb[4];

  // ---- prologue: batched loads for k0 = 0 (all issued before any wait) ----
#pragma unroll
  for (int i = 0; i < 4; ++i) {
    ra[i] = *(const float4*)(Abase + i * astep);
    rb[i] = *(const float4*)(Wbase + i * wstep);
  }

  int p = 0;
  for (int k0 = 0; k0 < K; k0 += BK, p ^= 1) {
    // ---- convert + store current registers into LDS buffer p ----
#pragma unroll
    for (int i = 0; i < 4; ++i) {
      float4 t = ra[i];
      if (PRE >= 1) {
        t.x = fmaf(t.x, scale[k0 + kq + 0], shift[k0 + kq + 0]);
        t.y = fmaf(t.y, scale[k0 + kq + 1], shift[k0 + kq + 1]);
        t.z = fmaf(t.z, scale[k0 + kq + 2], shift[k0 + kq + 2]);
        t.w = fmaf(t.w, scale[k0 + kq + 3], shift[k0 + kq + 3]);
      }
      if (PRE == 2) {
        t.x = fmaxf(t.x, 0.f); t.y = fmaxf(t.y, 0.f);
        t.z = fmaxf(t.z, 0.f); t.w = fmaxf(t.w, 0.f);
      }
      v4bf pka = { (bf16)t.x, (bf16)t.y, (bf16)t.z, (bf16)t.w };
      *(v4bf*)&lds_a[p][(r0 + i * 32) * BKP + kq] = pka;
      float4 w4 = rb[i];
      v4bf pkb = { (bf16)w4.x, (bf16)w4.y, (bf16)w4.z, (bf16)w4.w };
      *(v4bf*)&lds_b[p][(r0 + i * 32) * BKP + kq] = pkb;
    }
    __syncthreads();  // single barrier: ping-pong buffers make this safe

    // ---- issue next tile's global loads early (overlap with WMMA) ----
    const int kn = k0 + BK;
    if (kn < K) {
#pragma unroll
      for (int i = 0; i < 4; ++i) {
        const float* pa = Abase + kn + i * astep;
        const float* pw = Wbase + kn + i * wstep;
        ra[i] = *(const float4*)pa;
        rb[i] = *(const float4*)pw;
        __builtin_prefetch(pa + BK, 0, 1);  // speculative: faults dropped
        __builtin_prefetch(pw + BK, 0, 1);
      }
    }

    // ---- fragments per ISA 7.12.2 wave32 layouts ----
    Frag16 afrag[2], bfrag[4];
#pragma unroll
    for (int mi = 0; mi < 2; ++mi) {
      int r = wm * 32 + mi * 16 + lane_lo;
      const bf16* pf = &lds_a[p][r * BKP + lane_hi * 8];  // lanes16-31: K+8
      afrag[mi].h[0] = *(const v8bf*)pf;                  // K 0..7  (8..15)
      afrag[mi].h[1] = *(const v8bf*)(pf + 16);           // K 16..23 (24..31)
    }
#pragma unroll
    for (int ni = 0; ni < 4; ++ni) {
      int c = wn * 64 + ni * 16 + lane_lo;
      const bf16* pf = &lds_b[p][c * BKP + lane_hi * 8];  // lanes16-31: K+8
      bfrag[ni].h[0] = *(const v8bf*)pf;
      bfrag[ni].h[1] = *(const v8bf*)(pf + 16);
    }
#pragma unroll
    for (int mi = 0; mi < 2; ++mi)
#pragma unroll
      for (int ni = 0; ni < 4; ++ni)
        acc[mi][ni] = __builtin_amdgcn_wmma_f32_16x16x32_bf16(
            false, afrag[mi].v, false, bfrag[ni].v,
            (short)0, acc[mi][ni], false, false);
  }

  // ---- epilogue: bias, store, optional fused squared-error ----
  float lsum = 0.0f;
#pragma unroll
  for (int mi = 0; mi < 2; ++mi) {
#pragma unroll
    for (int ni = 0; ni < 4; ++ni) {
      int colg = col_blk + wn * 64 + ni * 16 + lane_lo;
      float bv = bias ? bias[colg] : 0.0f;
#pragma unroll
      for (int r = 0; r < 8; ++r) {
        int rowg = row_blk + wm * 32 + mi * 16 + lane_hi * 8 + r;
        float y  = acc[mi][ni][r] + bv;
        Y[(long)rowg * ldy + colg] = y;
        if (LOSS) {
          float d = y - Xref[(long)rowg * (long)Ncols + colg];
          lsum = fmaf(d, d, lsum);
        }
      }
    }
  }
  if (LOSS) {
    red[tid] = lsum;
    __syncthreads();
    for (int s = 128; s > 0; s >>= 1) {
      if (tid < s) red[tid] += red[tid + s];
      __syncthreads();
    }
    if (tid == 0) lossPartial[blockIdx.y * gridDim.x + blockIdx.x] = red[0];
  }
}

// ---------------------------------------------------------------------------
// Column partial stats (sum, sum-of-squares) over row chunks. Deterministic,
// no atomics: partials[chunk*Ncols + c].
// ---------------------------------------------------------------------------
__global__ __launch_bounds__(256) void k_colstats(const float* __restrict__ Y,
                                                  int Ncols, int rowsPerChunk,
                                                  float* __restrict__ psum,
                                                  float* __restrict__ psq) {
  int c     = blockIdx.x * 256 + threadIdx.x;
  int chunk = blockIdx.y;
  long base = (long)chunk * rowsPerChunk * Ncols + c;
  float s = 0.f, q = 0.f;
  for (int r = 0; r < rowsPerChunk; ++r) {
    float v = Y[base + (long)r * Ncols];
    s += v;
    q = fmaf(v, v, q);
  }
  psum[chunk * Ncols + c] = s;
  psq [chunk * Ncols + c] = q;
}

// BN finalize: scale = g/sqrt(var+eps), shift = beta - mean*scale
__global__ __launch_bounds__(256) void k_finalize_bn(
    const float* __restrict__ psum, const float* __restrict__ psq, int Ncols,
    int nchunk, float invN, const float* __restrict__ g,
    const float* __restrict__ be, float* __restrict__ scale,
    float* __restrict__ shift) {
  int c = blockIdx.x * 256 + threadIdx.x;
  if (c >= Ncols) return;
  float s = 0.f, q = 0.f;
  for (int i = 0; i < nchunk; ++i) { s += psum[i * Ncols + c]; q += psq[i * Ncols + c]; }
  float m   = s * invN;
  float var = q * invN - m * m;
  float sc  = g[c] * rsqrtf(var + EPSB);
  scale[c] = sc;
  shift[c] = be[c] - m * sc;
}

// ---------------------------------------------------------------------------
// Gumbel + dual argmax. One wave32 per (n,q): raw argmax (indices output) and
// noisy argmax over logits*exp(-log_temp) + gumbel(u) (hard code index).
// ---------------------------------------------------------------------------
__global__ __launch_bounds__(256) void k_gumbel_argmax(
    const float* __restrict__ Lraw, const float* __restrict__ u,
    const float* __restrict__ log_temps, float* __restrict__ idx_out,
    int* __restrict__ idx_hard) {
  int gwave = (blockIdx.x * 256 + threadIdx.x) >> 5;
  int lane  = threadIdx.x & 31;
  int n = gwave >> 3, q = gwave & 7;
  float ts   = __expf(-log_temps[q]);
  long base  = (long)n * (QN * CN) + q * CN;
  float bR = -3.4e38f, bN = -3.4e38f;
  int iR = 0, iN = 0;
#pragma unroll
  for (int j = 0; j < 8; ++j) {
    int c    = j * 32 + lane;
    float lr = Lraw[base + c];
    float uu = fminf(fmaxf(u[base + c], 1e-9f), 1.0f);
    float gn = -__logf(-__logf(uu) + 1e-20f);
    float nz = fmaf(lr, ts, gn);
    if (lr > bR) { bR = lr; iR = c; }   // c ascending within lane
    if (nz > bN) { bN = nz; iN = c; }
  }
  for (int off = 16; off > 0; off >>= 1) {
    float ovR = __shfl_xor(bR, off, 32); int oiR = __shfl_xor(iR, off, 32);
    float ovN = __shfl_xor(bN, off, 32); int oiN = __shfl_xor(iN, off, 32);
    if (ovR > bR || (ovR == bR && oiR < iR)) { bR = ovR; iR = oiR; }
    if (ovN > bN || (ovN == bN && oiN < iN)) { bN = ovN; iN = oiN; }
  }
  if (lane == 0) {
    idx_out[n * QN + q]  = (float)iR;
    idx_hard[n * QN + q] = iN;
  }
}

// codes @ codebook == codebook gather at hard index, summed over q
__global__ __launch_bounds__(256) void k_gather_emb(
    const float* __restrict__ codebook, const int* __restrict__ idx_hard,
    float* __restrict__ emb) {
  int n = blockIdx.x, d = threadIdx.x;
  float s = 0.f;
#pragma unroll
  for (int q = 0; q < QN; ++q) {
    int idx = idx_hard[n * QN + q];
    s += codebook[((long)q * CN + idx) * DN + d];
  }
  emb[(long)n * DN + d] = s;
}

__global__ __launch_bounds__(512) void k_loss_finalize(
    const float* __restrict__ partial, int nPart, float invCnt,
    float* __restrict__ out) {
  __shared__ float red[512];
  float s = 0.f;
  for (int i = threadIdx.x; i < nPart; i += 512) s += partial[i];
  red[threadIdx.x] = s;
  __syncthreads();
  for (int st = 256; st > 0; st >>= 1) {
    if (threadIdx.x < st) red[threadIdx.x] += red[threadIdx.x + st];
    __syncthreads();
  }
  if (threadIdx.x == 0) out[0] = red[0] * invCnt;
}

// ---------------------------------------------------------------------------
extern "C" void kernel_launch(void* const* d_in, const int* in_sizes, int n_in,
                              void* d_out, int out_size, void* d_ws,
                              size_t ws_size, hipStream_t stream) {
  const float* x      = (const float*)d_in[0];
  const float* u      = (const float*)d_in[1];
  const float* cb     = (const float*)d_in[2];
  const float* ltemps = (const float*)d_in[3];
  const float* enc_w0 = (const float*)d_in[4];
  const float* enc_b0 = (const float*)d_in[5];
  const float* enc_g0 = (const float*)d_in[6];
  const float* enc_be0= (const float*)d_in[7];
  const float* enc_w1 = (const float*)d_in[8];
  const float* enc_b1 = (const float*)d_in[9];
  const float* enc_g1 = (const float*)d_in[10];
  const float* enc_be1= (const float*)d_in[11];
  const float* enc_wo = (const float*)d_in[12];
  const float* enc_bo = (const float*)d_in[13];
  const float* dec_g  = (const float*)d_in[14];
  const float* dec_be = (const float*)d_in[15];
  const float* dec_w0 = (const float*)d_in[16];
  const float* dec_b0 = (const float*)d_in[17];
  const float* dec_g0 = (const float*)d_in[18];
  const float* dec_be0= (const float*)d_in[19];
  const float* dec_w1 = (const float*)d_in[20];
  const float* dec_b1 = (const float*)d_in[21];
  const float* dec_g1 = (const float*)d_in[22];
  const float* dec_be1= (const float*)d_in[23];
  const float* fin_w  = (const float*)d_in[24];
  const float* fin_b  = (const float*)d_in[25];

  float* out = (float*)d_out;
  float* x_reco  = out;                        // N*512
  float* idx_out = out + (size_t)NROWS * DIMX; // N*8
  float* loss_o  = idx_out + (size_t)NROWS * QN;

  // workspace layout (floats)
  float* ws = (float*)d_ws;
  size_t o = 0;
  float* xt   = ws + o; o += (size_t)NROWS * DIMX;       // transposed input
  float* Y1   = ws + o; o += (size_t)NROWS * HIDX;       // enc h0 pre-BN (reused: dec h0)
  float* Y2   = ws + o; o += (size_t)NROWS * HIDX;       // enc h1 pre-BN (reused: dec h1)
  float* Z    = ws + o; o += (size_t)NROWS * QN * DN;    // encoder output
  float* Lraw = ws + o; o += (size_t)NROWS * QN * CN;    // raw logits
  float* emb  = ws + o; o += (size_t)NROWS * DN;         // quantized embedding
  float* psum = ws + o; o += 32 * 2048;
  float* psq  = ws + o; o += 32 * 2048;
  float* bnSc = ws + o; o += 2048;
  float* bnSh = ws + o; o += 2048;
  int*   idxH = (int*)(ws + o); o += (size_t)NROWS * QN;
  float* lossP= ws + o; o += 512;

  const dim3 blk(256);
  const float invN = 1.0f / (float)NROWS;
  const int RC = 32, rowsPC = NROWS / RC;  // stat chunks

  // 1) transpose x -> xt
  k_transpose<<<dim3(512 / 32, DIMX / 32, 32), blk, 0, stream>>>(x, xt);

  // 2) encoder layer 0
  k_gemm_wmma<0, false><<<dim3(HIDX / 128, NROWS / 128), blk, 0, stream>>>(
      xt, DIMX, nullptr, nullptr, enc_w0, enc_b0, Y1, HIDX, DIMX, HIDX,
      nullptr, nullptr);
  k_colstats<<<dim3(HIDX / 256, RC), blk, 0, stream>>>(Y1, HIDX, rowsPC, psum, psq);
  k_finalize_bn<<<dim3(HIDX / 256), blk, 0, stream>>>(psum, psq, HIDX, RC, invN,
                                                      enc_g0, enc_be0, bnSc, bnSh);
  // 3) encoder layer 1 (fused BN+ReLU on A)
  k_gemm_wmma<2, false><<<dim3(HIDX / 128, NROWS / 128), blk, 0, stream>>>(
      Y1, HIDX, bnSc, bnSh, enc_w1, enc_b1, Y2, HIDX, HIDX, HIDX, nullptr, nullptr);
  k_colstats<<<dim3(HIDX / 256, RC), blk, 0, stream>>>(Y2, HIDX, rowsPC, psum, psq);
  k_finalize_bn<<<dim3(HIDX / 256), blk, 0, stream>>>(psum, psq, HIDX, RC, invN,
                                                      enc_g1, enc_be1, bnSc, bnSh);
  // 4) encoder output projection -> Z [N, Q*D]
  k_gemm_wmma<2, false><<<dim3((QN * DN) / 128, NROWS / 128), blk, 0, stream>>>(
      Y2, HIDX, bnSc, bnSh, enc_wo, enc_bo, Z, QN * DN, HIDX, QN * DN,
      nullptr, nullptr);

  // 5) per-codebook logits: Lraw[:, q, :] = Z[:, q, :] @ codebook[q]^T
  for (int q = 0; q < QN; ++q) {
    k_gemm_wmma<0, false><<<dim3(CN / 128, NROWS / 128), blk, 0, stream>>>(
        Z + q * DN, QN * DN, nullptr, nullptr, cb + (size_t)q * CN * DN,
        nullptr, Lraw + q * CN, QN * CN, DN, CN, nullptr, nullptr);
  }

  // 6) gumbel noise + dual argmax
  k_gumbel_argmax<<<dim3((NROWS * QN) / 8), blk, 0, stream>>>(Lraw, u, ltemps,
                                                              idx_out, idxH);
  // 7) embedding gather (codes @ codebook == hard one-hot gather)
  k_gather_emb<<<dim3(NROWS), blk, 0, stream>>>(cb, idxH, emb);

  // 8) decoder input BN
  k_colstats<<<dim3(DN / 256, RC), blk, 0, stream>>>(emb, DN, rowsPC, psum, psq);
  k_finalize_bn<<<dim3(1), blk, 0, stream>>>(psum, psq, DN, RC, invN, dec_g,
                                             dec_be, bnSc, bnSh);
  // 9) decoder layer 0 (BN on A, no relu)
  k_gemm_wmma<1, false><<<dim3(HIDX / 128, NROWS / 128), blk, 0, stream>>>(
      emb, DN, bnSc, bnSh, dec_w0, dec_b0, Y1, HIDX, DN, HIDX, nullptr, nullptr);
  k_colstats<<<dim3(HIDX / 256, RC), blk, 0, stream>>>(Y1, HIDX, rowsPC, psum, psq);
  k_finalize_bn<<<dim3(HIDX / 256), blk, 0, stream>>>(psum, psq, HIDX, RC, invN,
                                                      dec_g0, dec_be0, bnSc, bnSh);
  // 10) decoder layer 1
  k_gemm_wmma<2, false><<<dim3(HIDX / 128, NROWS / 128), blk, 0, stream>>>(
      Y1, HIDX, bnSc, bnSh, dec_w1, dec_b1, Y2, HIDX, HIDX, HIDX, nullptr, nullptr);
  k_colstats<<<dim3(HIDX / 256, RC), blk, 0, stream>>>(Y2, HIDX, rowsPC, psum, psq);
  k_finalize_bn<<<dim3(HIDX / 256), blk, 0, stream>>>(psum, psq, HIDX, RC, invN,
                                                      dec_g1, dec_be1, bnSc, bnSh);
  // 11) final projection + fused squared-error partials
  k_gemm_wmma<2, true><<<dim3(DIMX / 128, NROWS / 128), blk, 0, stream>>>(
      Y2, HIDX, bnSc, bnSh, fin_w, fin_b, x_reco, DIMX, HIDX, DIMX, xt, lossP);
  // 12) commit loss
  k_loss_finalize<<<dim3(1), dim3(512), 0, stream>>>(
      lossP, (DIMX / 128) * (NROWS / 128), 1.0f / ((float)NROWS * DIMX), loss_o);
}
